// MultiHeadedAttention_8718783611627
// MI455X (gfx1250) — compile-verified, matrix operand order checked by reading, not run on hardware
//
#include <hip/hip_runtime.h>
#include <hip/hip_bf16.h>

// ---------------------------------------------------------------------------
// MultiHeadedAttention forward for MI455X (gfx1250), wave32 + WMMA bf16.
//   B=2 S=2048 D=1024 H=16 DK=64
// Pipeline: cvt(f32->bf16) -> mask bitpack -> 3x proj GEMM (WMMA) ->
//           flash attention (async K/V staging to LDS + WMMA scores +
//           WMMA P*V, online softmax) -> out-proj GEMM (WMMA, f32 out).
// ---------------------------------------------------------------------------

typedef __bf16 bf16;
typedef __attribute__((ext_vector_type(16))) __bf16 v16bf;
typedef __attribute__((ext_vector_type(8)))  __bf16 v8bf;
typedef __attribute__((ext_vector_type(4)))  __bf16 v4bf;
typedef __attribute__((ext_vector_type(8)))  float  v8f;
typedef __attribute__((ext_vector_type(4)))  float  v4f;

#define WMMA_BF16(a, b, c) \
  __builtin_amdgcn_wmma_f32_16x16x32_bf16(false, (a), false, (b), (short)0, (c), false, false)

#define ZERO8 {0.f,0.f,0.f,0.f,0.f,0.f,0.f,0.f}

static constexpr int cB  = 2;
static constexpr int cS  = 2048;
static constexpr int cD  = 1024;
static constexpr int cH  = 16;
static constexpr int cDK = 64;
static constexpr int cM  = cB * cS;   // 4096 rows

// Load one 16-bit A/B WMMA fragment half: 8 elems at p, 8 elems at p+16
// (ISA 7.12.2: lane holds K = kb..kb+7 and kb+16..kb+23). Both are 16B loads.
__device__ __forceinline__ v16bf load_frag(const bf16* p) {
  v8bf lo = *(const v8bf*)(p);
  v8bf hi = *(const v8bf*)(p + 16);
  return __builtin_shufflevector(lo, hi, 0,1,2,3,4,5,6,7,8,9,10,11,12,13,14,15);
}

// Async 16-byte copy global -> LDS (ASYNCcnt path, GV addressing).
// VDST = LDS byte address (relative to wave LDS base), VADDR = 64-bit address.
__device__ __forceinline__ void async_copy_b128(unsigned lds_off, const bf16* g) {
  asm volatile("global_load_async_to_lds_b128 %0, %1, off"
               :: "v"(lds_off), "v"((unsigned long long)(uintptr_t)g)
               : "memory");
}

// ---------------------------------------------------------------------------
// f32 -> bf16 conversion (vector x4)
// ---------------------------------------------------------------------------
__global__ void cvt_f32_to_bf16(const float* __restrict__ in,
                                bf16* __restrict__ out, int n4) {
  int i = blockIdx.x * blockDim.x + threadIdx.x;
  if (i >= n4) return;
  v4f x = *(const v4f*)(in + (size_t)i * 4);
  v4bf y;
  y.x = (bf16)x.x; y.y = (bf16)x.y; y.z = (bf16)x.z; y.w = (bf16)x.w;
  *(v4bf*)(out + (size_t)i * 4) = y;
}

// ---------------------------------------------------------------------------
// Pack int32 mask [B,1,S,S] -> bitmask, 1 u32 per 32 keys.
// ---------------------------------------------------------------------------
__global__ void pack_mask(const int* __restrict__ mask,
                          unsigned* __restrict__ bits, int nwords) {
  int w = blockIdx.x * blockDim.x + threadIdx.x;
  if (w >= nwords) return;
  const int* p = mask + (size_t)w * 32;
  unsigned v = 0u;
  #pragma unroll
  for (int j = 0; j < 32; ++j) v |= (p[j] != 0) ? (1u << j) : 0u;
  bits[w] = v;
}

// ---------------------------------------------------------------------------
// GEMM: Y[M x 1024] = X[M x 1024] * W^T + bias.  X, W row-major bf16.
// B[k][n] = W[n][k] -> B fragment reads row n of W contiguously.
// One wave computes a 64x64 tile (4x4 of 16x16 WMMA). Block = 4 waves = 128x128.
// X (8MB) and W (2MB) stay L2-resident (192MB), so tile re-reads hit L2.
// ---------------------------------------------------------------------------
enum { LAY_QK = 0, LAY_VT = 1, LAY_F32 = 2 };

template <int LAYOUT>
__global__ __launch_bounds__(128) void gemm_xwT(
    const bf16* __restrict__ X, const bf16* __restrict__ Wr,
    const float* __restrict__ bias, void* __restrict__ out) {
  const int wave = threadIdx.x >> 5;
  const int lane = threadIdx.x & 31;
  const int m0 = blockIdx.x * 128 + (wave >> 1) * 64;
  const int n0 = blockIdx.y * 128 + (wave & 1) * 64;
  const int mr = lane & 15;        // A-row / B-col within 16-tile
  const int half = lane >> 4;
  const int kb = half * 8;         // per-lane K base (ISA A/B 16-bit layout)

  v8f acc[4][4];
  #pragma unroll
  for (int i = 0; i < 4; ++i)
    #pragma unroll
    for (int j = 0; j < 4; ++j) { v8f z = ZERO8; acc[i][j] = z; }

  for (int k0 = 0; k0 < cD; k0 += 32) {
    v16bf afr[4], bfr[4];
    #pragma unroll
    for (int i = 0; i < 4; ++i)
      afr[i] = load_frag(X + (size_t)(m0 + 16 * i + mr) * cD + k0 + kb);
    #pragma unroll
    for (int j = 0; j < 4; ++j)
      bfr[j] = load_frag(Wr + (size_t)(n0 + 16 * j + mr) * cD + k0 + kb);
    if (k0 + 32 < cD) {  // speculative L2 prefetch of next K-slab
      __builtin_prefetch(X  + (size_t)(m0 + mr) * cD + k0 + 32 + kb, 0, 0);
      __builtin_prefetch(Wr + (size_t)(n0 + mr) * cD + k0 + 32 + kb, 0, 0);
    }
    #pragma unroll
    for (int i = 0; i < 4; ++i)
      #pragma unroll
      for (int j = 0; j < 4; ++j)
        acc[i][j] = WMMA_BF16(afr[i], bfr[j], acc[i][j]);
  }

  // Epilogue. C layout: VGPR r, lanes 0-15 -> M=r, lanes 16-31 -> M=8+r.
  #pragma unroll
  for (int j = 0; j < 4; ++j) {
    const int n = n0 + 16 * j + mr;
    const float bn = bias[n];
    #pragma unroll
    for (int i = 0; i < 4; ++i) {
      #pragma unroll
      for (int r = 0; r < 8; ++r) {
        const float val = acc[i][j][r] + bn;
        const int m = m0 + 16 * i + r + 8 * half;
        if constexpr (LAYOUT == LAY_F32) {
          ((float*)out)[(size_t)m * cD + n] = val;
        } else {
          const int b = m >> 11, s = m & 2047;   // m = b*S + s
          const int h = n >> 6,  dk = n & 63;    // n = h*DK + dk
          if constexpr (LAYOUT == LAY_QK)        // [B,H,S,DK]
            ((bf16*)out)[(((size_t)(b * cH + h) * cS + s) * cDK + dk)] = (bf16)val;
          else                                   // [B,H,DK,S] (V transposed)
            ((bf16*)out)[(((size_t)(b * cH + h) * cDK + dk) * cS + s)] = (bf16)val;
        }
      }
    }
  }
}

// ---------------------------------------------------------------------------
// Flash attention. One wave per 16-row Q tile; 32 keys per chunk.
// K/V tiles for each chunk are staged into LDS with async global->LDS copies
// (double buffered, ASYNCcnt pipelined); WMMA fragments then come from LDS.
//  scores: 2 key-tiles x 2 WMMAs (K=DK=64); softmax online (shfl width-16);
//  P relayout C->A via per-wave LDS bounce; P*V: 4 WMMAs from Vt [B,H,DK,S].
// Block = 2 waves (64 threads): 2 x (2x4KB K + 2x4KB V + 1.25KB P) ~= 34.5KB LDS.
// ---------------------------------------------------------------------------
__global__ __launch_bounds__(64) void flash_attn(
    const bf16* __restrict__ Qh, const bf16* __restrict__ Kh,
    const bf16* __restrict__ Vt, const unsigned* __restrict__ mbits,
    bf16* __restrict__ Ctx) {
  __shared__ bf16 kstage[2][2][32 * 64];  // [wave][buf][key][dk]
  __shared__ bf16 vstage[2][2][64 * 32];  // [wave][buf][dk][key]
  __shared__ bf16 pstage[2][16 * 40];     // [wave][row][key(+pad)]

  const int wave = threadIdx.x >> 5;
  const int lane = threadIdx.x & 31;
  const int tile = blockIdx.x * 2 + wave;  // 0..4095
  const int qt = tile & 127;               // S/16 q-tiles
  const int bh = tile >> 7;                // b*H + h
  const int b = bh >> 4, h = bh & 15;
  const int q0 = qt * 16;
  const int mr = lane & 15, half = lane >> 4, kb = half * 8;

  const bf16* Kbh = Kh + (size_t)bh * cS * cDK;  // [key][dk]
  const bf16* Vbh = Vt + (size_t)bh * cDK * cS;  // [dk][key]

  // Stage one 32-key chunk: K tile 32x64 (4KB) + V tile 64x32 (4KB).
  // 256 16B segments each; lane handles segments lane, lane+32, ...
  // => 16 async instructions per wave per chunk.
  auto stage_kv = [&](int buf, int k0) {
    const unsigned kbase = (unsigned)(uintptr_t)&kstage[wave][buf][0];
    const unsigned vbase = (unsigned)(uintptr_t)&vstage[wave][buf][0];
    #pragma unroll
    for (int i = 0; i < 8; ++i) {
      const int seg = lane + 32 * i;
      const int key = seg >> 3, dk8 = seg & 7;          // K: 8 segs per row
      async_copy_b128(kbase + (unsigned)(key * 128 + dk8 * 16),
                      Kbh + (size_t)(k0 + key) * cDK + dk8 * 8);
      const int dkr = seg >> 2, k8 = seg & 3;           // V: 4 segs per row
      async_copy_b128(vbase + (unsigned)(dkr * 64 + k8 * 16),
                      Vbh + (size_t)dkr * cS + k0 + k8 * 8);
    }
  };

  // Q A-fragments: row q0+mr, two K-chunks (dk 0-31, 32-63)
  const bf16* Qbase = Qh + ((size_t)bh * cS + q0 + mr) * cDK;
  v16bf qf[2];
  qf[0] = load_frag(Qbase + kb);
  qf[1] = load_frag(Qbase + 32 + kb);

  float mrow[8], lrow[8];
  v8f oacc[4];
  #pragma unroll
  for (int r = 0; r < 8; ++r) { mrow[r] = -1e30f; lrow[r] = 0.f; }
  #pragma unroll
  for (int t = 0; t < 4; ++t) { v8f z = ZERO8; oacc[t] = z; }

  bf16* myp = &pstage[wave][0];
  const unsigned* mbase = mbits + ((size_t)b * cS + q0) * (cS / 32);

  stage_kv(0, 0);  // prologue: chunk 0 in flight

  for (int c = 0; c < cS / 32; ++c) {
    const int k0 = c * 32;
    const bf16* kc = &kstage[wave][c & 1][0];
    const bf16* vc = &vstage[wave][c & 1][0];

    if (c + 1 < cS / 32) {
      stage_kv((c + 1) & 1, k0 + 32);  // overlap next chunk's DMA
      // async ops complete in order per wave: <=16 outstanding means the
      // previous chunk (this iteration's data) has landed in LDS.
      asm volatile("s_wait_asynccnt 0x10" ::: "memory");
    } else {
      asm volatile("s_wait_asynccnt 0x0" ::: "memory");
    }

    // ---- scores for two 16-key tiles (fragments from LDS)
    v8f s[2];
    #pragma unroll
    for (int t = 0; t < 2; ++t) {
      v8f z = ZERO8; s[t] = z;
      const bf16* Kt = kc + (16 * t + mr) * 64;       // row stride 64 elems
      v16bf kf0 = load_frag(Kt + kb);
      v16bf kf1 = load_frag(Kt + 32 + kb);
      s[t] = WMMA_BF16(qf[0], kf0, s[t]);
      s[t] = WMMA_BF16(qf[1], kf1, s[t]);
    }

    // ---- scale + mask + online softmax (per row r; row = r + 8*half)
    const int kw = k0 >> 5;
    #pragma unroll
    for (int r = 0; r < 8; ++r) {
      const unsigned wb = mbase[(size_t)(r + 8 * half) * (cS / 32) + kw];
      float s0 = s[0][r] * 0.125f;             // 1/sqrt(64)
      float s1 = s[1][r] * 0.125f;
      if (!((wb >> mr) & 1u))        s0 = -1e9f;
      if (!((wb >> (16 + mr)) & 1u)) s1 = -1e9f;
      float vmax = fmaxf(s0, s1);
      #pragma unroll
      for (int d = 1; d < 16; d <<= 1) vmax = fmaxf(vmax, __shfl_xor(vmax, d, 16));
      const float mn = fmaxf(mrow[r], vmax);
      const float alpha = __expf(mrow[r] - mn);
      const float p0 = __expf(s0 - mn);
      const float p1 = __expf(s1 - mn);
      float psum = p0 + p1;
      #pragma unroll
      for (int d = 1; d < 16; d <<= 1) psum += __shfl_xor(psum, d, 16);
      lrow[r] = lrow[r] * alpha + psum;
      mrow[r] = mn;
      #pragma unroll
      for (int t = 0; t < 4; ++t) oacc[t][r] *= alpha;
      // stage P into LDS in [row][key] order for A-fragment reload
      const int row = r + 8 * half;
      myp[row * 40 + mr]      = (bf16)p0;
      myp[row * 40 + 16 + mr] = (bf16)p1;
    }
    asm volatile("s_wait_dscnt 0x0" ::: "memory");  // LDS RAW fence (same wave)

    // ---- reload P as a 16x32 A-fragment
    v16bf pf = load_frag(myp + mr * 40 + kb);
    // ---- P*V: B fragments from staged transposed V (row stride 32 elems)
    #pragma unroll
    for (int t = 0; t < 4; ++t) {
      v16bf vf = load_frag(vc + (16 * t + mr) * 32 + kb);
      oacc[t] = WMMA_BF16(pf, vf, oacc[t]);
    }
  }

  // ---- normalize + write context [B,S,D] bf16
  #pragma unroll
  for (int r = 0; r < 8; ++r) {
    const float inv = (lrow[r] > 0.f) ? (1.f / lrow[r]) : 0.f;
    const int q = q0 + r + 8 * half;
    #pragma unroll
    for (int t = 0; t < 4; ++t)
      Ctx[((size_t)b * cS + q) * cD + h * cDK + 16 * t + mr] =
          (bf16)(oacc[t][r] * inv);
  }
}

// ---------------------------------------------------------------------------
extern "C" void kernel_launch(void* const* d_in, const int* in_sizes, int n_in,
                              void* d_out, int out_size, void* d_ws, size_t ws_size,
                              hipStream_t stream) {
  const float* q   = (const float*)d_in[0];
  const float* k   = (const float*)d_in[1];
  const float* v   = (const float*)d_in[2];
  const int*   msk = (const int*)  d_in[3];
  const float* Wq  = (const float*)d_in[4];
  const float* bq  = (const float*)d_in[5];
  const float* Wk  = (const float*)d_in[6];
  const float* bk  = (const float*)d_in[7];
  const float* Wv  = (const float*)d_in[8];
  const float* bv  = (const float*)d_in[9];
  const float* Wo  = (const float*)d_in[10];
  const float* bo  = (const float*)d_in[11];

  const size_t nBSD = (size_t)cB * cS * cD;   // 4,194,304
  const size_t nDD  = (size_t)cD * cD;        // 1,048,576
  const int nMaskW  = cB * cS * (cS / 32);    // 262,144 words

  char* ws = (char*)d_ws;
  size_t off = 0;
  auto take = [&](size_t bytes) -> char* {
    char* p = ws + off;
    off += (bytes + 255) & ~(size_t)255;
    return p;
  };
  bf16* Xq  = (bf16*)take(nBSD * 2);
  bf16* Xk  = (bf16*)take(nBSD * 2);
  bf16* Xv  = (bf16*)take(nBSD * 2);
  bf16* Wqb = (bf16*)take(nDD * 2);
  bf16* Wkb = (bf16*)take(nDD * 2);
  bf16* Wvb = (bf16*)take(nDD * 2);
  bf16* Wob = (bf16*)take(nDD * 2);
  bf16* Qh  = (bf16*)take(nBSD * 2);
  bf16* Kh  = (bf16*)take(nBSD * 2);
  bf16* Vt  = (bf16*)take(nBSD * 2);
  bf16* Ctx = (bf16*)take(nBSD * 2);
  unsigned* Mb = (unsigned*)take((size_t)nMaskW * 4);

  // 1) conversions
  const int n4x = (int)(nBSD / 4), n4w = (int)(nDD / 4);
  cvt_f32_to_bf16<<<(n4x + 255) / 256, 256, 0, stream>>>(q,  Xq,  n4x);
  cvt_f32_to_bf16<<<(n4x + 255) / 256, 256, 0, stream>>>(k,  Xk,  n4x);
  cvt_f32_to_bf16<<<(n4x + 255) / 256, 256, 0, stream>>>(v,  Xv,  n4x);
  cvt_f32_to_bf16<<<(n4w + 255) / 256, 256, 0, stream>>>(Wq, Wqb, n4w);
  cvt_f32_to_bf16<<<(n4w + 255) / 256, 256, 0, stream>>>(Wk, Wkb, n4w);
  cvt_f32_to_bf16<<<(n4w + 255) / 256, 256, 0, stream>>>(Wv, Wvb, n4w);
  cvt_f32_to_bf16<<<(n4w + 255) / 256, 256, 0, stream>>>(Wo, Wob, n4w);
  // 2) mask bitpack
  pack_mask<<<(nMaskW + 255) / 256, 256, 0, stream>>>(msk, Mb, nMaskW);
  // 3) projections (WMMA)
  dim3 gg(cM / 128, cD / 128);  // (32, 8)
  gemm_xwT<LAY_QK><<<gg, 128, 0, stream>>>(Xq, Wqb, bq, Qh);
  gemm_xwT<LAY_QK><<<gg, 128, 0, stream>>>(Xk, Wkb, bk, Kh);
  gemm_xwT<LAY_VT><<<gg, 128, 0, stream>>>(Xv, Wvb, bv, Vt);
  // 4) attention (WMMA flash, async K/V staging)
  flash_attn<<<(cB * cH * (cS / 16)) / 2, 64, 0, stream>>>(Qh, Kh, Vt, Mb, Ctx);
  // 5) output projection (WMMA, f32 out + bias)
  gemm_xwT<LAY_F32><<<gg, 128, 0, stream>>>(Ctx, Wob, bo, d_out);
}
